// GNN_43447889166384
// MI455X (gfx1250) — matrix operand order repaired and checked by reading, not compiled, not run on hardware
//
#include <hip/hip_runtime.h>
#include <hip/hip_bf16.h>

#define Bq   32
#define Nq   1024
#define Hq   64
#define Lq   3
#define OUTq 2
#define NEGS 0.01f
#define Mq   (Bq * Nq)        // 32768 rows for the dense linears

typedef __attribute__((ext_vector_type(16))) __bf16 v16bf;
typedef __attribute__((ext_vector_type(8)))  __bf16 v8bf;
typedef __attribute__((ext_vector_type(8)))  float  v8f;

__device__ __forceinline__ float leaky(float x) { return x > 0.f ? x : NEGS * x; }

// A-operand fragment loader: row-major bf16, 16x32 tile, this code's own k order (not AMD's table order):
// lane = 16*half + r holds row r, K = half*16..+15 (contiguous)
__device__ __forceinline__ v16bf load_afrag(const __bf16* p) {
  return *(const v16bf*)p;
}

// ---------------------------------------------------------------------------
// deg: dis[b,n] = rsqrt(max(sum_m adj[b,n,m], 1e-12))   (one wave per row)
// ---------------------------------------------------------------------------
__global__ __launch_bounds__(256) void deg_kernel(const float* __restrict__ adj,
                                                  float* __restrict__ dis) {
  const int tid = threadIdx.x, lane = tid & 31;
  const int w = blockIdx.x * 8 + (tid >> 5);          // 0..32767
  const int b = w >> 10, n = w & 1023;
  const float* row = adj + ((size_t)b * Nq + n) * Nq;
  float s = 0.f;
  for (int m = lane; m < Nq; m += 32) s += row[m];
  for (int off = 16; off; off >>= 1) s += __shfl_xor(s, off, 32);
  if (lane == 0) dis[b * Nq + n] = rsqrtf(fmaxf(s, 1e-12f));
}

// ---------------------------------------------------------------------------
// Abf[b,n,m] = bf16( dis[b,n] * adj[b,n,m] * dis[b,m] )
// ---------------------------------------------------------------------------
__global__ __launch_bounds__(256) void norm_kernel(const float* __restrict__ adj,
                                                   const float* __restrict__ dis,
                                                   __bf16* __restrict__ Abf) {
  const size_t total4 = (size_t)Bq * Nq * Nq / 4;
  for (size_t i = (size_t)blockIdx.x * blockDim.x + threadIdx.x; i < total4;
       i += (size_t)gridDim.x * blockDim.x) {
    size_t e = i * 4;
    int b = (int)(e >> 20);
    int n = (int)((e >> 10) & 1023);
    int m0 = (int)(e & 1023);
    float4 v  = ((const float4*)adj)[i];
    float  dn = dis[(b << 10) + n];
    float4 dm = ((const float4*)(dis + (b << 10)))[m0 >> 2];
    Abf[e + 0] = (__bf16)(dn * v.x * dm.x);
    Abf[e + 1] = (__bf16)(dn * v.y * dm.y);
    Abf[e + 2] = (__bf16)(dn * v.z * dm.z);
    Abf[e + 3] = (__bf16)(dn * v.w * dm.w);
  }
}

// ---------------------------------------------------------------------------
// in_proj: h0 = X @ W^T + b ; writes f32 row, bf16 transposed, Xcat slice
// ---------------------------------------------------------------------------
__global__ __launch_bounds__(256) void prep_in_kernel(const float* __restrict__ X,
                                                      const float* __restrict__ w,
                                                      const float* __restrict__ bias,
                                                      float* __restrict__ Xf,
                                                      __bf16* __restrict__ XT,
                                                      __bf16* __restrict__ Xcat) {
  int idx = blockIdx.x * 256 + threadIdx.x;            // over Mq*Hq
  if (idx >= Mq * Hq) return;
  int m = idx >> 6, f = idx & 63;
  float x0 = X[m * 2], x1 = X[m * 2 + 1];
  float v = x0 * w[f * 2] + x1 * w[f * 2 + 1] + bias[f];
  Xf[idx] = v;
  Xcat[(size_t)m * 256 + f] = (__bf16)v;
  XT[(size_t)(m >> 10) * (Hq * Nq) + (size_t)f * Nq + (m & 1023)] = (__bf16)v;
}

// ---------------------------------------------------------------------------
// f32 -> bf16 copy (weight prep)
// ---------------------------------------------------------------------------
__global__ void cvt_kernel(const float* __restrict__ src, __bf16* __restrict__ dst, int n) {
  for (int i = blockIdx.x * blockDim.x + threadIdx.x; i < n; i += gridDim.x * blockDim.x)
    dst[i] = (__bf16)src[i];
}

// ---------------------------------------------------------------------------
// Hop GEMM: out = A_norm(bf16,[N,N]) @ h(bf16T,[H,N])   per batch.
// One wave -> 32 rows x 64 cols (two A frags share each B frag from LDS:
// 8 WMMAs per 8 ds_b128 + 4 global_b128 per K-step). 8 waves/block -> 256
// rows; grid (4, B). B-operand (h^T, 128 KB) staged entirely in LDS.
// Writes pre-activation bf16T (next hop input) and leaky-activated f32 rows.
// ---------------------------------------------------------------------------
__global__ __launch_bounds__(256) void spmm_kernel(const __bf16* __restrict__ A,
                                                   const __bf16* __restrict__ hT,
                                                   __bf16* __restrict__ outT,
                                                   float* __restrict__ outAct) {
  __shared__ __bf16 sB[Hq * Nq];                       // 128 KB
  const int b = blockIdx.y;
  const int tid = threadIdx.x, wave = tid >> 5, lane = tid & 31;
  {
    const uint4* src = (const uint4*)(hT + (size_t)b * Hq * Nq);
    uint4* dst = (uint4*)sB;
    for (int i = tid; i < Hq * Nq / 8; i += 256) dst[i] = src[i];
  }
  __syncthreads();

  const int rowBase = blockIdx.x * 256 + wave * 32;
  const int r = lane & 15, half = lane >> 4;
  const __bf16* arow0 = A + (size_t)b * Nq * Nq + (size_t)(rowBase + r) * Nq + half * 16;
  const __bf16* arow1 = arow0 + (size_t)16 * Nq;

  v8f acc[8] = {};
  for (int k = 0; k < Nq; k += 32) {
    v16bf af0 = load_afrag(arow0 + k);
    v16bf af1 = load_afrag(arow1 + k);
    // keep the HBM stream ahead: 512 B (~8 K-steps) of lookahead per row stream
    __builtin_prefetch(arow0 + k + 256, 0, 0);
    __builtin_prefetch(arow1 + k + 256, 0, 0);
    const __bf16* bb = sB + (size_t)r * Nq + k + half * 16;
#pragma unroll
    for (int nt = 0; nt < 4; ++nt) {
      v16bf bf = *(const v16bf*)(bb + nt * 16 * Nq);
      acc[nt]     = __builtin_amdgcn_wmma_f32_16x16x32_bf16(false, af0, false, bf, (short)0, acc[nt],     false, false);
      acc[4 + nt] = __builtin_amdgcn_wmma_f32_16x16x32_bf16(false, af1, false, bf, (short)0, acc[4 + nt], false, false);
    }
  }

  __bf16* oT = outT + (size_t)b * Hq * Nq;
  float*  oA = outAct + (size_t)b * Nq * Hq;
#pragma unroll
  for (int t = 0; t < 2; ++t) {
#pragma unroll
    for (int nt = 0; nt < 4; ++nt) {
      const int c = nt * 16 + r;
#pragma unroll
      for (int rr = 0; rr < 8; ++rr) {
        const int m = rowBase + t * 16 + rr + half * 8;
        float v = acc[t * 4 + nt][rr];
        oT[(size_t)c * Nq + m] = (__bf16)v;            // pre-activation
        oA[(size_t)m * Hq + c] = leaky(v);             // activated
      }
    }
  }
}

// ---------------------------------------------------------------------------
// Attention over hop order + the reference's cat(dim=0).view batch remap,
// produces h_prime in bf16 rows. One wave per (b,n).
// ---------------------------------------------------------------------------
__global__ __launch_bounds__(256) void attn_kernel(const float* __restrict__ Xf,
                                                   const float* __restrict__ h1,
                                                   const float* __restrict__ h2,
                                                   const float* __restrict__ h3,
                                                   const float* __restrict__ a,
                                                   __bf16* __restrict__ hp) {
  const int tid = threadIdx.x, lane = tid & 31;
  const int w = blockIdx.x * 8 + (tid >> 5);           // 0..32767
  const int b = w >> 10, n = w & 1023;
  const size_t base = (size_t)w * Hq;
  float dh = 0.f, d1 = 0.f, d2 = 0.f, d3 = 0.f;
  for (int f = lane; f < Hq; f += 32) {
    float alo = a[f], ahi = a[Hq + f];
    dh += fmaxf(Xf[base + f], 0.f) * alo;
    d1 += fmaxf(h1[base + f], 0.f) * ahi;
    d2 += fmaxf(h2[base + f], 0.f) * ahi;
    d3 += fmaxf(h3[base + f], 0.f) * ahi;
  }
  for (int off = 16; off; off >>= 1) {
    dh += __shfl_xor(dh, off, 32);
    d1 += __shfl_xor(d1, off, 32);
    d2 += __shfl_xor(d2, off, 32);
    d3 += __shfl_xor(d3, off, 32);
  }
  float e1 = dh + d1, e2 = dh + d2, e3 = dh + d3;
  float mx = fmaxf(e1, fmaxf(e2, e3));
  float a1 = __expf(e1 - mx), a2 = __expf(e2 - mx), a3 = __expf(e3 - mx);
  float inv = 1.f / (3.f * (a1 + a2 + a3));            // softmax * mean(1/3)
  a1 *= inv; a2 *= inv; a3 *= inv;

  const float* p[3];
#pragma unroll
  for (int k = 0; k < 3; ++k) {                        // j=3b+k -> hop j/32, batch j%32
    int j = 3 * b + k;
    const float* src = (j >> 5) == 0 ? h1 : ((j >> 5) == 1 ? h2 : h3);
    p[k] = src + (((size_t)(j & 31) << 10) + n) * Hq;
  }
  for (int f = lane; f < Hq; f += 32) {
    float v = a1 * p[0][f] + a2 * p[1][f] + a3 * p[2][f];
    hp[base + f] = (__bf16)v;
  }
}

// ---------------------------------------------------------------------------
// Dense linear (WMMA): Y = leaky(X[M,K] @ W[64,K]^T + bias).
// Writes bf16 rows (stride/coloff for the Xcat slice), optional f32 rows,
// optional bf16 transposed [B][H][N]. One wave -> 16x64; grid 256 blocks.
// ---------------------------------------------------------------------------
__global__ __launch_bounds__(256) void linear_kernel(const __bf16* __restrict__ Xin, int K,
                                                     const __bf16* __restrict__ W,
                                                     const float* __restrict__ bias,
                                                     __bf16* __restrict__ outB, int outStride,
                                                     int outColOff,
                                                     float* __restrict__ outF,
                                                     __bf16* __restrict__ outT) {
  const int tid = threadIdx.x, wave = tid >> 5, lane = tid & 31;
  const int rowBase = blockIdx.x * 128 + wave * 16;
  const int r = lane & 15, half = lane >> 4;
  const __bf16* arow = Xin + (size_t)(rowBase + r) * K + half * 16;

  v8f acc0 = {}, acc1 = {}, acc2 = {}, acc3 = {};
  for (int k = 0; k < K; k += 32) {
    v16bf af = load_afrag(arow + k);
    const __bf16* wb = W + (size_t)r * K + k + half * 16;
    v16bf b0 = *(const v16bf*)(wb + 0  * 16 * K);
    v16bf b1 = *(const v16bf*)(wb + 1  * 16 * K);
    v16bf b2 = *(const v16bf*)(wb + 2  * 16 * K);
    v16bf b3 = *(const v16bf*)(wb + 3  * 16 * K);
    acc0 = __builtin_amdgcn_wmma_f32_16x16x32_bf16(false, af, false, b0, (short)0, acc0, false, false);
    acc1 = __builtin_amdgcn_wmma_f32_16x16x32_bf16(false, af, false, b1, (short)0, acc1, false, false);
    acc2 = __builtin_amdgcn_wmma_f32_16x16x32_bf16(false, af, false, b2, (short)0, acc2, false, false);
    acc3 = __builtin_amdgcn_wmma_f32_16x16x32_bf16(false, af, false, b3, (short)0, acc3, false, false);
  }

  v8f accs[4] = {acc0, acc1, acc2, acc3};
#pragma unroll
  for (int nt = 0; nt < 4; ++nt) {
    const int c = nt * 16 + r;
    const float bsv = bias[c];
#pragma unroll
    for (int rr = 0; rr < 8; ++rr) {
      const int m = rowBase + rr + half * 8;
      float v = leaky(accs[nt][rr] + bsv);
      outB[(size_t)m * outStride + outColOff + c] = (__bf16)v;
      if (outF) outF[(size_t)m * Hq + c] = v;
      if (outT)
        outT[(size_t)(m >> 10) * (Hq * Nq) + (size_t)c * Nq + (m & 1023)] = (__bf16)v;
    }
  }
}

// ---------------------------------------------------------------------------
// MLP head: logits = X[M,64] @ W[2,64]^T + b   (tiny, VALU)
// ---------------------------------------------------------------------------
__global__ __launch_bounds__(256) void mlp2_kernel(const __bf16* __restrict__ Xin,
                                                   const float* __restrict__ W,
                                                   const float* __restrict__ bias,
                                                   float* __restrict__ logits) {
  int m = blockIdx.x * 256 + threadIdx.x;
  if (m >= Mq) return;
  float a0 = bias[0], a1 = bias[1];
  const v8bf* xr = (const v8bf*)(Xin + (size_t)m * Hq);
#pragma unroll
  for (int g = 0; g < 8; ++g) {
    v8bf x = xr[g];
#pragma unroll
    for (int j = 0; j < 8; ++j) {
      float xv = (float)x[j];
      a0 += xv * W[g * 8 + j];
      a1 += xv * W[Hq + g * 8 + j];
    }
  }
  logits[(size_t)m * 2 + 0] = a0;
  logits[(size_t)m * 2 + 1] = a1;
}

// ---------------------------------------------------------------------------
// Softmax over nodes (axis=1). One wave per (b, o); 64 waves total.
// ---------------------------------------------------------------------------
__global__ __launch_bounds__(256) void softmax_nodes_kernel(const float* __restrict__ logits,
                                                            float* __restrict__ out) {
  const int tid = threadIdx.x, lane = tid & 31;
  const int w = blockIdx.x * 8 + (tid >> 5);           // 0..63
  const int b = w >> 1, o = w & 1;
  const float* src = logits + (size_t)b * Nq * OUTq + o;
  float mx = -1e30f;
  for (int n = lane; n < Nq; n += 32) mx = fmaxf(mx, src[(size_t)n * OUTq]);
  for (int off = 16; off; off >>= 1) mx = fmaxf(mx, __shfl_xor(mx, off, 32));
  float s = 0.f;
  for (int n = lane; n < Nq; n += 32) s += __expf(src[(size_t)n * OUTq] - mx);
  for (int off = 16; off; off >>= 1) s += __shfl_xor(s, off, 32);
  float inv = 1.f / s;
  for (int n = lane; n < Nq; n += 32)
    out[(size_t)b * Nq * OUTq + (size_t)n * OUTq + o] = __expf(src[(size_t)n * OUTq] - mx) * inv;
}

// ---------------------------------------------------------------------------
extern "C" void kernel_launch(void* const* d_in, const int* in_sizes, int n_in,
                              void* d_out, int out_size, void* d_ws, size_t ws_size,
                              hipStream_t stream) {
  const float* X        = (const float*)d_in[0];
  const float* adj      = (const float*)d_in[1];
  const float* inw      = (const float*)d_in[2];
  const float* inb      = (const float*)d_in[3];
  const float* lin1_w   = (const float*)d_in[4];
  const float* lin1_b   = (const float*)d_in[5];
  const float* lin2_w   = (const float*)d_in[6];
  const float* lin2_b   = (const float*)d_in[7];
  const float* att_a    = (const float*)d_in[8];
  const float* mlp1_w   = (const float*)d_in[9];
  const float* mlp1_b   = (const float*)d_in[10];
  const float* mlp2_w   = (const float*)d_in[11];
  const float* mlp2_b   = (const float*)d_in[12];

  char* ws = (char*)d_ws;
  size_t off = 0;
  auto take = [&](size_t bytes) { char* p = ws + off; off += (bytes + 255) & ~(size_t)255; return p; };

  __bf16* Abf   = (__bf16*)take((size_t)Bq * Nq * Nq * 2);    // 64 MB normalized adj
  float*  dis   = (float*) take((size_t)Bq * Nq * 4);
  __bf16* XT    = (__bf16*)take((size_t)Bq * Hq * Nq * 2);    // layer input, transposed
  __bf16* hTa   = (__bf16*)take((size_t)Bq * Hq * Nq * 2);    // hop ping
  __bf16* hTb   = (__bf16*)take((size_t)Bq * Hq * Nq * 2);    // hop pong
  float*  Xf    = (float*) take((size_t)Mq * Hq * 4);         // layer input, f32 rows
  float*  h1f   = (float*) take((size_t)Mq * Hq * 4);
  float*  h2f   = (float*) take((size_t)Mq * Hq * 4);
  float*  h3f   = (float*) take((size_t)Mq * Hq * 4);
  __bf16* hpB   = (__bf16*)take((size_t)Mq * Hq * 2);         // h_prime bf16 rows
  __bf16* l1B   = (__bf16*)take((size_t)Mq * Hq * 2);
  __bf16* Xcat  = (__bf16*)take((size_t)Mq * 256 * 2);        // hidden concat, bf16
  __bf16* m1B   = (__bf16*)take((size_t)Mq * Hq * 2);
  float*  lgt   = (float*) take((size_t)Mq * OUTq * 4);
  __bf16* Wl1   = (__bf16*)take((size_t)Lq * Hq * Hq * 2);
  __bf16* Wl2   = (__bf16*)take((size_t)Lq * Hq * Hq * 2);
  __bf16* Wm1   = (__bf16*)take((size_t)Hq * 256 * 2);
  (void)ws_size; (void)in_sizes; (void)n_in; (void)out_size;

  // --- one-time (per call) prep ---
  deg_kernel<<<4096, 256, 0, stream>>>(adj, dis);
  norm_kernel<<<8192, 256, 0, stream>>>(adj, dis, Abf);
  prep_in_kernel<<<(Mq * Hq) / 256, 256, 0, stream>>>(X, inw, inb, Xf, XT, Xcat);
  cvt_kernel<<<64, 256, 0, stream>>>(lin1_w, Wl1, Lq * Hq * Hq);
  cvt_kernel<<<64, 256, 0, stream>>>(lin2_w, Wl2, Lq * Hq * Hq);
  cvt_kernel<<<64, 256, 0, stream>>>(mlp1_w, Wm1, Hq * 256);

  // --- 3 GCN layers ---
  for (int l = 0; l < Lq; ++l) {
    dim3 g(Nq / 256, Bq);
    spmm_kernel<<<g, 256, 0, stream>>>(Abf, XT,  hTa, h1f);   // h1 = A @ X
    spmm_kernel<<<g, 256, 0, stream>>>(Abf, hTa, hTb, h2f);   // h2 = A @ h1
    spmm_kernel<<<g, 256, 0, stream>>>(Abf, hTb, hTa, h3f);   // h3 = A @ h2
    attn_kernel<<<4096, 256, 0, stream>>>(Xf, h1f, h2f, h3f, att_a + l * 2 * Hq, hpB);
    linear_kernel<<<256, 256, 0, stream>>>(hpB, Hq, Wl1 + l * Hq * Hq, lin1_b + l * Hq,
                                           l1B, Hq, 0, nullptr, nullptr);
    linear_kernel<<<256, 256, 0, stream>>>(l1B, Hq, Wl2 + l * Hq * Hq, lin2_b + l * Hq,
                                           Xcat, 256, Hq * (l + 1), Xf, XT);
  }

  // --- MLP head + node softmax ---
  linear_kernel<<<256, 256, 0, stream>>>(Xcat, 256, Wm1, mlp1_b, m1B, Hq, 0, nullptr, nullptr);
  mlp2_kernel<<<Mq / 256, 256, 0, stream>>>(m1B, mlp2_w, mlp2_b, lgt);
  softmax_nodes_kernel<<<8, 256, 0, stream>>>(lgt, (float*)d_out);
}
